// Decoder_21844203667929
// MI455X (gfx1250) — matrix operand order repaired and checked by reading, not compiled, not run on hardware
//
#include <hip/hip_runtime.h>
#include <hip/hip_bf16.h>

#define T_DIM 512
#define B_DIM 128
#define I_DIM 512
#define H_DIM 512
#define O_DIM 12
#define L_DIM 16

typedef __attribute__((ext_vector_type(16))) __bf16 v16bf;
typedef __attribute__((ext_vector_type(8)))  __bf16 v8bf;
typedef __attribute__((ext_vector_type(8)))  float  v8f;
typedef __attribute__((ext_vector_type(4))) unsigned int v4ui;
typedef __attribute__((ext_vector_type(8))) int          v8si;
typedef __attribute__((ext_vector_type(4))) int          v4si;

#if defined(__HIP_DEVICE_COMPILE__) && __has_builtin(__builtin_amdgcn_tensor_load_to_lds)
#define HAVE_TDM 1
#else
#define HAVE_TDM 0
#endif

// ---------------------------------------------------------------------------
// Tensor Data Mover: DMA a 2-D tile (tile_w x tile_h elems, row stride
// stride_elems) from global into LDS at lds_off.  dsz_code: 0=1B,1=2B,2=4B.
// Caller must __syncthreads() after (issue from a single wave).
// ---------------------------------------------------------------------------
__device__ __forceinline__ void tdm_load_2d(const void* g, unsigned int lds_off,
                                            unsigned int tile_w, unsigned int tile_h,
                                            unsigned int stride_elems,
                                            unsigned int dsz_code,
                                            unsigned int tensor_w) {
#if HAVE_TDM
  unsigned long long ga = (unsigned long long)g;
  v4ui g0;
  g0.x = 1u;                                            // count=1, user mode
  g0.y = lds_off;                                       // lds_addr
  g0.z = (unsigned int)ga;                              // global_addr[31:0]
  g0.w = (unsigned int)((ga >> 32) & 0x01FFFFFFull) | (2u << 30);  // hi + type=2
  v8si g1;
  g1[0] = (int)(dsz_code << 16);                        // data_size
  g1[1] = (int)((tensor_w & 0xFFFFu) << 16);            // tensor_dim0 lo16
  g1[2] = (int)((tensor_w >> 16) & 0xFFFFu);            // tensor_dim0 hi16 (dim1 lo16=0)
  g1[3] = (int)(16u | (tile_w << 16));                  // tensor_dim1=1<<20, tile_dim0
  g1[4] = (int)tile_h;                                  // tile_dim1
  g1[5] = (int)stride_elems;                            // tensor_dim0_stride lo32
  g1[6] = 0;
  g1[7] = 0;
  v4si z4 = {0, 0, 0, 0};
#if __clang_major__ >= 23
  v8si z8 = {0, 0, 0, 0, 0, 0, 0, 0};
  __builtin_amdgcn_tensor_load_to_lds(g0, g1, z4, z4, z8, 0);
#else
  __builtin_amdgcn_tensor_load_to_lds(g0, g1, z4, z4, 0);
#endif
#if __has_builtin(__builtin_amdgcn_s_wait_tensorcnt)
  __builtin_amdgcn_s_wait_tensorcnt(0);
#else
  asm volatile("s_wait_tensorcnt 0x0" ::: "memory");
#endif
#else
  (void)g; (void)lds_off; (void)tile_w; (void)tile_h;
  (void)stride_elems; (void)dsz_code; (void)tensor_w;
#endif
}

// ---------------------------------------------------------------------------
// f32 -> bf16 bulk convert (x tensor), 4 elems/thread, packed 8B stores
// ---------------------------------------------------------------------------
__global__ void k_f32_to_bf16(const float* __restrict__ src,
                              __bf16* __restrict__ dst, int n) {
  int idx = (blockIdx.x * blockDim.x + threadIdx.x) * 4;
  if (idx + 3 < n) {
    float4 v = *(const float4*)(src + idx);
    union { short4 s4; __bf16 h[4]; } u;
    u.h[0] = (__bf16)v.x; u.h[1] = (__bf16)v.y;
    u.h[2] = (__bf16)v.z; u.h[3] = (__bf16)v.w;
    *(short4*)(dst + idx) = u.s4;
  }
}

// ---------------------------------------------------------------------------
// Generic bf16-WMMA GEMM:  C[M,N] (+)= A[M,K] @ B[K,N]
//   block = 256 thr (8 waves), tile 32(M) x 64(N), K-chunk 32
//   A tile LDS row-major (TDM-loaded when A is bf16); B tile transposed,
//   rows padded to 40 elems.  Fragments = v8bf (ds_load_b128) pairs.
//   Output staged through LDS for coalesced b128 stores.
// ---------------------------------------------------------------------------
template <typename TA, int ACT, bool ACC, bool OUT_BF16>
__global__ __launch_bounds__(256)
void k_gemm_wmma(const TA* __restrict__ A, int lda,
                 const float* __restrict__ Bm, int ldb,
                 float* __restrict__ Cf, __bf16* __restrict__ Cb,
                 int ldc, int K) {
  __shared__ union SM {
    struct { __bf16 a[32 * 32]; __bf16 bt[64 * 40]; } st;
    float  of[32 * 64];
    __bf16 ob[32 * 64];
  } sm;

  const int tid  = threadIdx.x;
  const int lane = tid & 31;
  const int wave = tid >> 5;
  const int m0 = blockIdx.y * 32;
  const int n0 = blockIdx.x * 64;
  const int wm = (wave & 1) * 16;
  const int wn = (wave >> 1) * 16;
  const int hi = lane >> 4;
  const int lm = lane & 15;
  constexpr bool A_IS_BF16 = (sizeof(TA) == 2);

  v8f acc = {};

  for (int kc = 0; kc < K; kc += 32) {
    // ---- stage A tile (32 rows x 32 k), row-major bf16 ----
    if constexpr (A_IS_BF16) {
#if HAVE_TDM
      if (wave == 0)
        tdm_load_2d(A + (size_t)m0 * lda + kc,
                    (unsigned int)(size_t)(&sm.st.a[0]),
                    32u, 32u, (unsigned int)lda, 1u, (unsigned int)K);
#else
      for (int p = tid; p < 512; p += 256) {
        int m = p >> 4, kp = (p & 15) * 2;
        const unsigned int* au =
            (const unsigned int*)(A + (size_t)(m0 + m) * lda + kc + kp);
        ((unsigned int*)sm.st.a)[(m * 32 + kp) >> 1] = *au;
      }
#endif
    } else {
      for (int p = tid; p < 512; p += 256) {
        int m = p >> 4, kp = (p & 15) * 2;
        float2 f2 = *(const float2*)(A + (size_t)(m0 + m) * lda + kc + kp);
        union { unsigned int u; __bf16 h[2]; } pk;
        pk.h[0] = (__bf16)f2.x; pk.h[1] = (__bf16)f2.y;
        ((unsigned int*)sm.st.a)[(m * 32 + kp) >> 1] = pk.u;
      }
    }

    // ---- stage B tile transposed: Bs_t[n][k], row stride 40 elems ----
    for (int p = tid; p < 1024; p += 256) {
      int c = p & 63, r2 = p >> 6;                  // r2 = k-pair index
      float f0 = Bm[(size_t)(kc + 2 * r2) * ldb + n0 + c];
      float f1 = Bm[(size_t)(kc + 2 * r2 + 1) * ldb + n0 + c];
      union { unsigned int u; __bf16 h[2]; } pk;
      pk.h[0] = (__bf16)f0; pk.h[1] = (__bf16)f1;
      ((unsigned int*)sm.st.bt)[c * 20 + r2] = pk.u;
    }
    __syncthreads();

    // ---- fragments: v8bf (b128) loads concatenated into v16bf ----
    const int am = (wm + lm) * 32 + hi * 8;
    v8bf a_lo = *(const v8bf*)&sm.st.a[am];         // K = hi*8 .. +7
    v8bf a_hi = *(const v8bf*)&sm.st.a[am + 16];    // K = 16+hi*8 .. +7
    v16bf fa = __builtin_shufflevector(a_lo, a_hi, 0, 1, 2, 3, 4, 5, 6, 7,
                                       8, 9, 10, 11, 12, 13, 14, 15);
    const int bn = (wn + lm) * 40 + hi * 8;
    v8bf b_lo = *(const v8bf*)&sm.st.bt[bn];        // K = hi*8 .. +7
    v8bf b_hi = *(const v8bf*)&sm.st.bt[bn + 16];   // K = 16+hi*8 .. +7
    v16bf fb = __builtin_shufflevector(b_lo, b_hi, 0, 1, 2, 3, 4, 5, 6, 7,
                                       8, 9, 10, 11, 12, 13, 14, 15);
    acc = __builtin_amdgcn_wmma_f32_16x16x32_bf16(
        false, fa, false, fb, (short)0, acc, false, false);
    __syncthreads();
  }

  // ---- output: stage D tile in LDS, coalesced writeback ----
  if constexpr (OUT_BF16) {
#pragma unroll
    for (int v = 0; v < 8; ++v) {
      float val = acc[v];
      if (ACT == 1) val = tanhf(val);
      sm.ob[(wm + v + 8 * hi) * 64 + wn + lm] = (__bf16)val;
    }
    __syncthreads();
    int row = tid >> 3, col = (tid & 7) * 8;        // 8 bf16 = 16B per thread
    float4 d = *(const float4*)&sm.ob[row * 64 + col];
    *(float4*)(Cb + (size_t)(m0 + row) * ldc + n0 + col) = d;
  } else {
#pragma unroll
    for (int v = 0; v < 8; ++v) {
      float val = acc[v];
      if (ACT == 1) val = tanhf(val);
      sm.of[(wm + v + 8 * hi) * 64 + wn + lm] = val;
    }
    __syncthreads();
    int row = tid >> 3, col = (tid & 7) * 8;        // 8 f32 = 32B per thread
    float* gp = Cf + (size_t)(m0 + row) * ldc + n0 + col;
    float4 d0 = *(const float4*)&sm.of[row * 64 + col];
    float4 d1 = *(const float4*)&sm.of[row * 64 + col + 4];
    if (ACC) {
      float4 c0v = *(const float4*)gp;
      float4 c1v = *(const float4*)(gp + 4);
      d0.x += c0v.x; d0.y += c0v.y; d0.z += c0v.z; d0.w += c0v.w;
      d1.x += c1v.x; d1.y += c1v.y; d1.z += c1v.z; d1.w += c1v.w;
    }
    *(float4*)gp = d0;
    *(float4*)(gp + 4) = d1;
  }
}

// ---------------------------------------------------------------------------
// e[b,t] = sum_h va[h] * tanh(sWa[b,h] + U[t,b,h])  -- one wave per (t,b)
// (e stored transposed [B][T] for contiguous softmax/context access)
// ---------------------------------------------------------------------------
__global__ __launch_bounds__(256)
void k_attn_e(const __bf16* __restrict__ U, const float* __restrict__ sWa,
              const float* __restrict__ va, float* __restrict__ e_out) {
  int wave = threadIdx.x >> 5, lane = threadIdx.x & 31;
  int g = blockIdx.x * 8 + wave;          // g = t*B + b
  int b = g & (B_DIM - 1);
  int t = g >> 7;
  const __bf16* Urow = U + (size_t)g * H_DIM;
  const float*  srow = sWa + (size_t)b * H_DIM;
  float p = 0.f;
  for (int h = lane * 4; h < H_DIM; h += 128) {
    float4 sv = *(const float4*)(srow + h);
    float4 vv = *(const float4*)(va + h);
    union { float2 f2; __bf16 hh[4]; } uu;
    uu.f2 = *(const float2*)(Urow + h);
    p += vv.x * tanhf(sv.x + (float)uu.hh[0]);
    p += vv.y * tanhf(sv.y + (float)uu.hh[1]);
    p += vv.z * tanhf(sv.z + (float)uu.hh[2]);
    p += vv.w * tanhf(sv.w + (float)uu.hh[3]);
  }
#pragma unroll
  for (int off = 16; off > 0; off >>= 1) p += __shfl_xor(p, off, 32);
  if (lane == 0) e_out[(size_t)b * T_DIM + t] = p;
}

// ---------------------------------------------------------------------------
// softmax over T (contiguous row per b), in place; one block per b
// ---------------------------------------------------------------------------
__global__ __launch_bounds__(256)
void k_softmax_t(float* __restrict__ e) {
  __shared__ float red[256];
  int b = blockIdx.x, tid = threadIdx.x;
  float* row = e + (size_t)b * T_DIM;
  float v0 = row[tid];
  float v1 = row[tid + 256];
  red[tid] = fmaxf(v0, v1);
  __syncthreads();
  for (int s = 128; s > 0; s >>= 1) {
    if (tid < s) red[tid] = fmaxf(red[tid], red[tid + s]);
    __syncthreads();
  }
  float mx = red[0];
  __syncthreads();
  float ex0 = expf(v0 - mx), ex1 = expf(v1 - mx);
  red[tid] = ex0 + ex1;
  __syncthreads();
  for (int s = 128; s > 0; s >>= 1) {
    if (tid < s) red[tid] += red[tid + s];
    __syncthreads();
  }
  float inv = 1.f / red[0];
  row[tid]       = ex0 * inv;
  row[tid + 256] = ex1 * inv;
}

// ---------------------------------------------------------------------------
// c[b,i] = sum_t a[b,t] * x[t,b,i]  -- one block (128 thr) per b, 4 i/thread
// attention row a[b,:] DMA'd into LDS via TDM (contiguous 1x512 f32 tile)
// ---------------------------------------------------------------------------
__global__ __launch_bounds__(128)
void k_context(const float* __restrict__ a, const __bf16* __restrict__ xb,
               float* __restrict__ c) {
  __shared__ float aw[T_DIM];
  int b = blockIdx.x;
#if HAVE_TDM
  if (threadIdx.x < 32)
    tdm_load_2d(a + (size_t)b * T_DIM, (unsigned int)(size_t)(&aw[0]),
                T_DIM, 1u, T_DIM, 2u, T_DIM);
#else
  for (int t = threadIdx.x; t < T_DIM; t += 128)
    aw[t] = a[(size_t)b * T_DIM + t];
#endif
  __syncthreads();
  int i0 = threadIdx.x * 4;
  float a0 = 0.f, a1 = 0.f, a2 = 0.f, a3 = 0.f;
  for (int t = 0; t < T_DIM; ++t) {
    union { float2 f2; __bf16 hh[4]; } uu;
    uu.f2 = *(const float2*)(xb + ((size_t)t * B_DIM + b) * I_DIM + i0);
    float w = aw[t];
    a0 += w * (float)uu.hh[0];
    a1 += w * (float)uu.hh[1];
    a2 += w * (float)uu.hh[2];
    a3 += w * (float)uu.hh[3];
  }
  float4 r; r.x = a0; r.y = a1; r.z = a2; r.w = a3;
  *(float4*)(c + (size_t)b * I_DIM + i0) = r;
}

// ---------------------------------------------------------------------------
// r = sigmoid(R + out@wr), z = sigmoid(Z + out@wz); emit rs = r*s and z
// ---------------------------------------------------------------------------
__global__ __launch_bounds__(256)
void k_gates_rz(const float* __restrict__ R, const float* __restrict__ Z,
                const float* __restrict__ outb, const float* __restrict__ wr,
                const float* __restrict__ wz, const float* __restrict__ s,
                float* __restrict__ rs, float* __restrict__ zb) {
  int idx = blockIdx.x * blockDim.x + threadIdx.x;  // over B*H
  int b = idx >> 9, h = idx & (H_DIM - 1);
  float dr = R[idx], dz = Z[idx];
#pragma unroll
  for (int o = 0; o < O_DIM; ++o) {
    float ov = outb[b * O_DIM + o];
    dr += ov * wr[o * H_DIM + h];
    dz += ov * wz[o * H_DIM + h];
  }
  float r = 1.f / (1.f + expf(-dr));
  float z = 1.f / (1.f + expf(-dz));
  rs[idx] = r * s[idx];
  zb[idx] = z;
}

// ---------------------------------------------------------------------------
// sh = tanh(G + out@w0); s = (1-z)*s + z*sh  (in place)
// ---------------------------------------------------------------------------
__global__ __launch_bounds__(256)
void k_update_s(const float* __restrict__ G, const float* __restrict__ outb,
                const float* __restrict__ w0, const float* __restrict__ zb,
                float* __restrict__ s) {
  int idx = blockIdx.x * blockDim.x + threadIdx.x;
  int b = idx >> 9, h = idx & (H_DIM - 1);
  float g = G[idx];
#pragma unroll
  for (int o = 0; o < O_DIM; ++o)
    g += outb[b * O_DIM + o] * w0[o * H_DIM + h];
  float sh = tanhf(g);
  float z = zb[idx];
  s[idx] = (1.f - z) * s[idx] + z * sh;
}

// ---------------------------------------------------------------------------
// out[b,o] = s[b,:] . fc_w[:,o] + fc_b[o]; one wave per (b,o), 12 waves/block
// ---------------------------------------------------------------------------
__global__ __launch_bounds__(384)
void k_out_proj(const float* __restrict__ s, const float* __restrict__ fc_w,
                const float* __restrict__ fc_b, float* __restrict__ outb,
                float* __restrict__ dout, int l) {
  int b = blockIdx.x;
  int o = threadIdx.x >> 5, lane = threadIdx.x & 31;
  float p = 0.f;
  for (int h = lane; h < H_DIM; h += 32)
    p += s[(size_t)b * H_DIM + h] * fc_w[h * O_DIM + o];
#pragma unroll
  for (int off = 16; off > 0; off >>= 1) p += __shfl_xor(p, off, 32);
  if (lane == 0) {
    float v = p + fc_b[o];
    outb[b * O_DIM + o] = v;
    dout[((size_t)b * L_DIM + l) * O_DIM + o] = v;
  }
}

// ---------------------------------------------------------------------------
extern "C" void kernel_launch(void* const* d_in, const int* in_sizes, int n_in,
                              void* d_out, int out_size, void* d_ws, size_t ws_size,
                              hipStream_t stream) {
  (void)in_sizes; (void)n_in; (void)out_size; (void)ws_size;
  const float* x    = (const float*)d_in[0];
  const float* w0   = (const float*)d_in[1];
  const float* wz   = (const float*)d_in[2];
  const float* wr   = (const float*)d_in[3];
  const float* wsm  = (const float*)d_in[4];
  const float* wa   = (const float*)d_in[5];
  const float* ua   = (const float*)d_in[6];
  const float* va   = (const float*)d_in[7];
  const float* u0   = (const float*)d_in[8];
  const float* uz   = (const float*)d_in[9];
  const float* ur   = (const float*)d_in[10];
  const float* c0   = (const float*)d_in[11];
  const float* cz   = (const float*)d_in[12];
  const float* cr   = (const float*)d_in[13];
  const float* fc_w = (const float*)d_in[14];
  const float* fc_b = (const float*)d_in[15];
  float* dout = (float*)d_out;

  char* ws = (char*)d_ws;
  size_t off = 0;
  auto carve = [&](size_t bytes) {
    char* p = ws + off;
    off = (off + bytes + 255) & ~(size_t)255;
    return p;
  };
  const size_t TBH = (size_t)T_DIM * B_DIM * H_DIM;
  __bf16* xb  = (__bf16*)carve(TBH * sizeof(__bf16));
  __bf16* Ub  = (__bf16*)carve(TBH * sizeof(__bf16));
  float*  e   = (float*)carve((size_t)T_DIM * B_DIM * 4);
  float*  sWa = (float*)carve((size_t)B_DIM * H_DIM * 4);
  float*  sbf = (float*)carve((size_t)B_DIM * H_DIM * 4);
  float*  cb  = (float*)carve((size_t)B_DIM * I_DIM * 4);
  float*  Rb  = (float*)carve((size_t)B_DIM * H_DIM * 4);
  float*  Zb  = (float*)carve((size_t)B_DIM * H_DIM * 4);
  float*  Gb  = (float*)carve((size_t)B_DIM * H_DIM * 4);
  float*  rsb = (float*)carve((size_t)B_DIM * H_DIM * 4);
  float*  zbf = (float*)carve((size_t)B_DIM * H_DIM * 4);
  float*  outb = (float*)carve((size_t)B_DIM * O_DIM * 4);

  const int nX = T_DIM * B_DIM * I_DIM;

  k_f32_to_bf16<<<nX / (256 * 4), 256, 0, stream>>>(x, xb, nX);

  {
    dim3 grid(H_DIM / 64, (T_DIM * B_DIM) / 32);
    k_gemm_wmma<__bf16, 0, false, true><<<grid, 256, 0, stream>>>(
        xb, I_DIM, ua, H_DIM, nullptr, Ub, H_DIM, I_DIM);
  }
  {
    dim3 grid(H_DIM / 64, B_DIM / 32);
    k_gemm_wmma<float, 1, false, false><<<grid, 256, 0, stream>>>(
        x, I_DIM, wsm, H_DIM, sbf, nullptr, H_DIM, I_DIM);
  }
  k_out_proj<<<B_DIM, 384, 0, stream>>>(sbf, fc_w, fc_b, outb, dout, 0);

  dim3 gBH(H_DIM / 64, B_DIM / 32);
  for (int l = 1; l < L_DIM; ++l) {
    k_gemm_wmma<float, 0, false, false><<<gBH, 256, 0, stream>>>(
        sbf, H_DIM, wa, H_DIM, sWa, nullptr, H_DIM, H_DIM);
    k_attn_e<<<(T_DIM * B_DIM) / 8, 256, 0, stream>>>(Ub, sWa, va, e);
    k_softmax_t<<<B_DIM, 256, 0, stream>>>(e);
    k_context<<<B_DIM, 128, 0, stream>>>(e, xb, cb);
    k_gemm_wmma<float, 0, false, false><<<gBH, 256, 0, stream>>>(
        sbf, H_DIM, ur, H_DIM, Rb, nullptr, H_DIM, H_DIM);
    k_gemm_wmma<float, 0, true, false><<<gBH, 256, 0, stream>>>(
        cb, I_DIM, cr, H_DIM, Rb, nullptr, H_DIM, I_DIM);
    k_gemm_wmma<float, 0, false, false><<<gBH, 256, 0, stream>>>(
        sbf, H_DIM, uz, H_DIM, Zb, nullptr, H_DIM, H_DIM);
    k_gemm_wmma<float, 0, true, false><<<gBH, 256, 0, stream>>>(
        cb, I_DIM, cz, H_DIM, Zb, nullptr, H_DIM, I_DIM);
    k_gates_rz<<<(B_DIM * H_DIM) / 256, 256, 0, stream>>>(
        Rb, Zb, outb, wr, wz, sbf, rsb, zbf);
    k_gemm_wmma<float, 0, false, false><<<gBH, 256, 0, stream>>>(
        rsb, H_DIM, u0, H_DIM, Gb, nullptr, H_DIM, H_DIM);
    k_gemm_wmma<float, 0, true, false><<<gBH, 256, 0, stream>>>(
        cb, I_DIM, c0, H_DIM, Gb, nullptr, H_DIM, I_DIM);
    k_update_s<<<(B_DIM * H_DIM) / 256, 256, 0, stream>>>(
        Gb, outb, w0, zbf, sbf);
    k_out_proj<<<B_DIM, 384, 0, stream>>>(sbf, fc_w, fc_b, outb, dout, l);
  }
}
